// Cell2Vec_12043088298541
// MI455X (gfx1250) — matrix operand order repaired and checked by reading, not compiled, not run on hardware
//
#include <hip/hip_runtime.h>

typedef __attribute__((ext_vector_type(16))) __bf16 v16bf;
typedef __attribute__((ext_vector_type(8)))  float  v8f;

#define N_NODES 50000
#define N_EDGES 400000
#define IN_F    512
#define HID     512
#define OUT_F   256
#define N_CELL  1000
#define N_DIM   128
#define N_SEL   8192
#define N_C     1024

// Native hardware convert (RNE) — lets clang emit packed bf16 cvt ops instead
// of a 3-VALU bit-manip sequence per element.
__device__ __forceinline__ __bf16 f2bf(float f) { return (__bf16)f; }

// ---------------------------------------------------------------------------
// Degree accumulation (fp32 hardware atomics) and rsqrt norms
// ---------------------------------------------------------------------------
__global__ __launch_bounds__(256) void degrees_kernel(
    const int* __restrict__ src, const int* __restrict__ dst,
    float* __restrict__ deg_out, float* __restrict__ deg_in, int E) {
  int e = blockIdx.x * blockDim.x + threadIdx.x;
  if (e >= E) return;
  unsafeAtomicAdd(&deg_out[src[e]], 1.0f);
  unsafeAtomicAdd(&deg_in[dst[e]], 1.0f);
}

__global__ __launch_bounds__(256) void norms_kernel(
    float* __restrict__ ns, float* __restrict__ nd, int n) {
  int i = blockIdx.x * blockDim.x + threadIdx.x;
  if (i >= n) return;
  float a = ns[i];
  ns[i] = (a > 0.f) ? rsqrtf(fmaxf(a, 1.f)) : 0.f;
  float b = nd[i];
  nd[i] = (b > 0.f) ? rsqrtf(fmaxf(b, 1.f)) : 0.f;
}

// ---------------------------------------------------------------------------
// Pack fp32 weight W[K][N] into per-lane WMMA-B bf16 layout:
//   Bp[((nt*nKt + kt)*32 + lane)*16 + j] = bf16( W[kt*32 + (lane&16) + j][nt*16 + (lane&15)] )
// so each lane's v16bf fragment is one contiguous 32-byte load.
// ---------------------------------------------------------------------------
__global__ __launch_bounds__(256) void pack_weight_kernel(
    const float* __restrict__ W, __bf16* __restrict__ Bp, int K, int N) {
  int nKt = K >> 5;
  int total = (N >> 4) * nKt * 32;
  int tid = blockIdx.x * blockDim.x + threadIdx.x;
  if (tid >= total) return;
  int lane = tid & 31;
  int kt = (tid >> 5) % nKt;
  int nt = (tid >> 5) / nKt;
  int n  = nt * 16 + (lane & 15);
  int kb = kt * 32 + (lane & 16);
  v16bf frag;
#pragma unroll
  for (int j = 0; j < 16; ++j)
    frag[j] = f2bf(W[(size_t)(kb + j) * N + n]);
  *(v16bf*)(Bp + (size_t)tid * 16) = frag;
}

// ---------------------------------------------------------------------------
// Edge scatter-add:  agg[dst] += h[src] * ns[src], 512 channels per edge.
// One 128-thread block per edge, float4 strips, hardware fp32 atomics.
// ---------------------------------------------------------------------------
__global__ __launch_bounds__(128) void edge_agg_kernel(
    const float* __restrict__ h, const int* __restrict__ src,
    const int* __restrict__ dst, const float* __restrict__ ns,
    float* __restrict__ agg, int E) {
  int e = blockIdx.x;
  if (e >= E) return;
  int s = src[e], d = dst[e];
  float w = ns[s];
  int c = threadIdx.x << 2;                       // 128 threads * 4 = 512 ch
  float4 v = *(const float4*)(h + (size_t)s * 512 + c);
  float* o = agg + (size_t)d * 512 + c;
  unsafeAtomicAdd(o + 0, v.x * w);
  unsafeAtomicAdd(o + 1, v.y * w);
  unsafeAtomicAdd(o + 2, v.z * w);
  unsafeAtomicAdd(o + 3, v.w * w);
}

// ---------------------------------------------------------------------------
// Build A fragment (16-bit A 16x32 layout) from a row-major fp32 row.
// Lane L: row = L&15 ; K offsets (L<16 ? 0 : 16)+[0..15].
// ---------------------------------------------------------------------------
__device__ __forceinline__ v16bf load_a_frag(const float* __restrict__ arow,
                                             int kt, int kbA, float scale) {
  const float4* p0 = (const float4*)(arow + kt * 32 + kbA);
  float4 a0 = p0[0], a1 = p0[1];
  const float4* p1 = (const float4*)(arow + kt * 32 + kbA + 8);
  float4 a2 = p1[0], a3 = p1[1];
  v16bf v;
  v[0]  = f2bf(a0.x * scale); v[1]  = f2bf(a0.y * scale);
  v[2]  = f2bf(a0.z * scale); v[3]  = f2bf(a0.w * scale);
  v[4]  = f2bf(a1.x * scale); v[5]  = f2bf(a1.y * scale);
  v[6]  = f2bf(a1.z * scale); v[7]  = f2bf(a1.w * scale);
  v[8]  = f2bf(a2.x * scale); v[9]  = f2bf(a2.y * scale);
  v[10] = f2bf(a2.z * scale); v[11] = f2bf(a2.w * scale);
  v[12] = f2bf(a3.x * scale); v[13] = f2bf(a3.y * scale);
  v[14] = f2bf(a3.z * scale); v[15] = f2bf(a3.w * scale);
  return v;
}

// ---------------------------------------------------------------------------
// GraphConv dense stage:  C = relu( (agg * nd_row) @ W + b )
// One wave per (M-tile 16) x (N-group 64). K multiple of 32.
// ---------------------------------------------------------------------------
__global__ __launch_bounds__(256) void gemm_layer_kernel(
    const float* __restrict__ A, const float* __restrict__ nd,
    const __bf16* __restrict__ Bp, const float* __restrict__ bias,
    float* __restrict__ C, int Mt, int ngroups, int K, int N) {
  int lane = threadIdx.x & 31;
  int wid = blockIdx.x * (blockDim.x >> 5) + (threadIdx.x >> 5);
  int mt = wid / ngroups, ng = wid % ngroups;
  if (mt >= Mt) return;

  int mrow = mt * 16 + (lane & 15);
  float scale = nd[mrow];
  int kbA = lane & 16;                            // 0 or 16
  int nKt = K >> 5;
  const float* arow = A + (size_t)mrow * K;
  size_t tstride = (size_t)nKt * 32 * 16;         // packed elems per N-tile

  v8f acc0 = {}, acc1 = {}, acc2 = {}, acc3 = {};
  for (int kt = 0; kt < nKt; ++kt) {
    v16bf af = load_a_frag(arow, kt, kbA, scale);
    const __bf16* bbase =
        Bp + (((size_t)(ng * 4) * nKt + (size_t)kt) * 32 + lane) * 16;
    v16bf b0 = *(const v16bf*)(bbase);
    v16bf b1 = *(const v16bf*)(bbase + tstride);
    v16bf b2 = *(const v16bf*)(bbase + 2 * tstride);
    v16bf b3 = *(const v16bf*)(bbase + 3 * tstride);
    acc0 = __builtin_amdgcn_wmma_f32_16x16x32_bf16(false, af, false, b0, (short)0, acc0, false, false);
    acc1 = __builtin_amdgcn_wmma_f32_16x16x32_bf16(false, af, false, b1, (short)0, acc1, false, false);
    acc2 = __builtin_amdgcn_wmma_f32_16x16x32_bf16(false, af, false, b2, (short)0, acc2, false, false);
    acc3 = __builtin_amdgcn_wmma_f32_16x16x32_bf16(false, af, false, b3, (short)0, acc3, false, false);
  }

  int rowoff = (lane & 16) >> 1;                  // +8 for hi-half lanes
  int ncol = lane & 15;
#pragma unroll
  for (int t = 0; t < 4; ++t) {
    v8f a = (t == 0) ? acc0 : (t == 1) ? acc1 : (t == 2) ? acc2 : acc3;
    int nglob = (ng * 4 + t) * 16 + ncol;
    float bv = bias[nglob];
#pragma unroll
    for (int r = 0; r < 8; ++r) {
      float v = a[r] + bv;
      v = v > 0.f ? v : 0.f;
      C[(size_t)(mt * 16 + r + rowoff) * N + nglob] = v;
    }
  }
}

// ---------------------------------------------------------------------------
// Projection: proj = h3[x_indices] @ Wp + bp  -> stored as bf16 [N_SEL][N_DIM]
// ---------------------------------------------------------------------------
__global__ __launch_bounds__(256) void gemm_proj_kernel(
    const float* __restrict__ H, const int* __restrict__ xind,
    const __bf16* __restrict__ Bp, const float* __restrict__ bias,
    __bf16* __restrict__ Cb, int Mt, int ngroups, int K, int N) {
  int lane = threadIdx.x & 31;
  int wid = blockIdx.x * (blockDim.x >> 5) + (threadIdx.x >> 5);
  int mt = wid / ngroups, ng = wid % ngroups;
  if (mt >= Mt) return;

  int row = xind[mt * 16 + (lane & 15)];
  int kbA = lane & 16;
  int nKt = K >> 5;
  const float* arow = H + (size_t)row * K;
  size_t tstride = (size_t)nKt * 32 * 16;

  v8f acc0 = {}, acc1 = {}, acc2 = {}, acc3 = {};
  for (int kt = 0; kt < nKt; ++kt) {
    v16bf af = load_a_frag(arow, kt, kbA, 1.0f);
    const __bf16* bbase =
        Bp + (((size_t)(ng * 4) * nKt + (size_t)kt) * 32 + lane) * 16;
    v16bf b0 = *(const v16bf*)(bbase);
    v16bf b1 = *(const v16bf*)(bbase + tstride);
    v16bf b2 = *(const v16bf*)(bbase + 2 * tstride);
    v16bf b3 = *(const v16bf*)(bbase + 3 * tstride);
    acc0 = __builtin_amdgcn_wmma_f32_16x16x32_bf16(false, af, false, b0, (short)0, acc0, false, false);
    acc1 = __builtin_amdgcn_wmma_f32_16x16x32_bf16(false, af, false, b1, (short)0, acc1, false, false);
    acc2 = __builtin_amdgcn_wmma_f32_16x16x32_bf16(false, af, false, b2, (short)0, acc2, false, false);
    acc3 = __builtin_amdgcn_wmma_f32_16x16x32_bf16(false, af, false, b3, (short)0, acc3, false, false);
  }

  int rowoff = (lane & 16) >> 1;
  int ncol = lane & 15;
#pragma unroll
  for (int t = 0; t < 4; ++t) {
    v8f a = (t == 0) ? acc0 : (t == 1) ? acc1 : (t == 2) ? acc2 : acc3;
    int nglob = (ng * 4 + t) * 16 + ncol;
    float bv = bias[nglob];
#pragma unroll
    for (int r = 0; r < 8; ++r)
      Cb[(size_t)(mt * 16 + r + rowoff) * N + nglob] = f2bf(a[r] + bv);
  }
}

// ---------------------------------------------------------------------------
// Final: out[c, s] = emb[c_indices[c], :] . proj[s, :]
// B fragment read straight out of bf16 proj rows (K contiguous per lane).
// ---------------------------------------------------------------------------
__global__ __launch_bounds__(256) void gemm_final_kernel(
    const float* __restrict__ EMB, const int* __restrict__ cind,
    const __bf16* __restrict__ Pb, float* __restrict__ out,
    int Mt, int ngroups, int K, int N) {
  int lane = threadIdx.x & 31;
  int wid = blockIdx.x * (blockDim.x >> 5) + (threadIdx.x >> 5);
  int mt = wid / ngroups, ng = wid % ngroups;
  if (mt >= Mt) return;

  int row = cind[mt * 16 + (lane & 15)];
  int kbA = lane & 16;
  int kbB = lane & 16;
  int ncol = lane & 15;
  int nKt = K >> 5;
  const float* arow = EMB + (size_t)row * K;

  v8f acc0 = {}, acc1 = {}, acc2 = {}, acc3 = {};
  for (int kt = 0; kt < nKt; ++kt) {
    v16bf af = load_a_frag(arow, kt, kbA, 1.0f);
    int n0 = (ng * 4) * 16 + ncol;
    const v16bf* pb0 = (const v16bf*)(Pb + (size_t)(n0)      * K + kt * 32 + kbB);
    const v16bf* pb1 = (const v16bf*)(Pb + (size_t)(n0 + 16) * K + kt * 32 + kbB);
    const v16bf* pb2 = (const v16bf*)(Pb + (size_t)(n0 + 32) * K + kt * 32 + kbB);
    const v16bf* pb3 = (const v16bf*)(Pb + (size_t)(n0 + 48) * K + kt * 32 + kbB);
    acc0 = __builtin_amdgcn_wmma_f32_16x16x32_bf16(false, af, false, *pb0, (short)0, acc0, false, false);
    acc1 = __builtin_amdgcn_wmma_f32_16x16x32_bf16(false, af, false, *pb1, (short)0, acc1, false, false);
    acc2 = __builtin_amdgcn_wmma_f32_16x16x32_bf16(false, af, false, *pb2, (short)0, acc2, false, false);
    acc3 = __builtin_amdgcn_wmma_f32_16x16x32_bf16(false, af, false, *pb3, (short)0, acc3, false, false);
  }

  int rowoff = (lane & 16) >> 1;
#pragma unroll
  for (int t = 0; t < 4; ++t) {
    v8f a = (t == 0) ? acc0 : (t == 1) ? acc1 : (t == 2) ? acc2 : acc3;
    int nglob = (ng * 4 + t) * 16 + ncol;
#pragma unroll
    for (int r = 0; r < 8; ++r)
      out[(size_t)(mt * 16 + r + rowoff) * N + nglob] = a[r];
  }
}

// ---------------------------------------------------------------------------
extern "C" void kernel_launch(void* const* d_in, const int* in_sizes, int n_in,
                              void* d_out, int out_size, void* d_ws, size_t ws_size,
                              hipStream_t stream) {
  const float* x   = (const float*)d_in[0];
  const int*   src = (const int*)d_in[1];
  const int*   dst = (const int*)d_in[2];
  const int*   xin = (const int*)d_in[3];
  const int*   cin = (const int*)d_in[4];
  const float* W1  = (const float*)d_in[5];
  const float* b1  = (const float*)d_in[6];
  const float* W2  = (const float*)d_in[7];
  const float* b2  = (const float*)d_in[8];
  const float* W3  = (const float*)d_in[9];
  const float* b3  = (const float*)d_in[10];
  const float* Wp  = (const float*)d_in[11];
  const float* bp  = (const float*)d_in[12];
  const float* emb = (const float*)d_in[13];
  float* out = (float*)d_out;

  char* ws = (char*)d_ws;
  size_t off = 0;
  auto take = [&](size_t bytes) -> char* {
    char* p = ws + off;
    off += (bytes + 255) & ~(size_t)255;
    return p;
  };
  float* agg = (float*)take((size_t)N_NODES * 512 * 4);
  float* h   = (float*)take((size_t)N_NODES * 512 * 4);
  float* ns  = (float*)take((size_t)N_NODES * 4);
  float* nd  = (float*)take((size_t)N_NODES * 4);
  __bf16* W1p = (__bf16*)take((size_t)512 * 512 * 2);
  __bf16* W2p = (__bf16*)take((size_t)512 * 512 * 2);
  __bf16* W3p = (__bf16*)take((size_t)512 * 256 * 2);
  __bf16* Wpp = (__bf16*)take((size_t)256 * 128 * 2);
  __bf16* Pb  = (__bf16*)take((size_t)N_SEL * N_DIM * 2);

  // --- degrees & norms ---
  hipMemsetAsync(ns, 0, (size_t)N_NODES * 4, stream);
  hipMemsetAsync(nd, 0, (size_t)N_NODES * 4, stream);
  degrees_kernel<<<(N_EDGES + 255) / 256, 256, 0, stream>>>(src, dst, ns, nd, N_EDGES);
  norms_kernel<<<(N_NODES + 255) / 256, 256, 0, stream>>>(ns, nd, N_NODES);

  // --- pack weights to bf16 WMMA-B layout ---
  {
    int t1 = (512 / 16) * (512 / 32) * 32;
    pack_weight_kernel<<<(t1 + 255) / 256, 256, 0, stream>>>(W1, W1p, 512, 512);
    pack_weight_kernel<<<(t1 + 255) / 256, 256, 0, stream>>>(W2, W2p, 512, 512);
    int t3 = (256 / 16) * (512 / 32) * 32;
    pack_weight_kernel<<<(t3 + 255) / 256, 256, 0, stream>>>(W3, W3p, 512, 256);
    int tp = (128 / 16) * (256 / 32) * 32;
    pack_weight_kernel<<<(tp + 255) / 256, 256, 0, stream>>>(Wp, Wpp, 256, 128);
  }

  const int Mt = N_NODES / 16;  // 3125 exact

  // --- layer 1: x -> h ---
  hipMemsetAsync(agg, 0, (size_t)N_NODES * 512 * 4, stream);
  edge_agg_kernel<<<N_EDGES, 128, 0, stream>>>(x, src, dst, ns, agg, N_EDGES);
  gemm_layer_kernel<<<(Mt * 8 + 7) / 8, 256, 0, stream>>>(agg, nd, W1p, b1, h, Mt, 8, 512, 512);

  // --- layer 2: h -> h ---
  hipMemsetAsync(agg, 0, (size_t)N_NODES * 512 * 4, stream);
  edge_agg_kernel<<<N_EDGES, 128, 0, stream>>>(h, src, dst, ns, agg, N_EDGES);
  gemm_layer_kernel<<<(Mt * 8 + 7) / 8, 256, 0, stream>>>(agg, nd, W2p, b2, h, Mt, 8, 512, 512);

  // --- layer 3: h -> h (256 wide) ---
  hipMemsetAsync(agg, 0, (size_t)N_NODES * 512 * 4, stream);
  edge_agg_kernel<<<N_EDGES, 128, 0, stream>>>(h, src, dst, ns, agg, N_EDGES);
  gemm_layer_kernel<<<(Mt * 4 + 7) / 8, 256, 0, stream>>>(agg, nd, W3p, b3, h, Mt, 4, 512, 256);

  // --- projection: proj(bf16) = h[x_indices] @ Wp + bp ---
  {
    int mtp = N_SEL / 16;  // 512
    gemm_proj_kernel<<<(mtp * 2 + 7) / 8, 256, 0, stream>>>(h, xin, Wpp, bp, Pb, mtp, 2, 256, 128);
  }

  // --- final: out = emb[c_indices] @ proj^T ---
  {
    int mtf = N_C / 16;          // 64
    int ngf = N_SEL / 64;        // 128
    gemm_final_kernel<<<(mtf * ngf + 7) / 8, 256, 0, stream>>>(emb, cin, Pb, out, mtf, ngf, 128, N_SEL);
  }
}
